// MoELayer_50422916055541
// MI455X (gfx1250) — compile-verified, matrix operand order checked by reading, not run on hardware
//
#include <hip/hip_runtime.h>

#define NB 8
#define SS 8192
#define DD 256
#define HH 1024
#define EE 8
#define NT (NB * SS)        // 65536 tokens
#define MTILE 128
#define HCHUNK 64
#define NCHUNK (HH / HCHUNK)
#define XS_STRIDE 264       // 256 + 8 pad (ushorts) -> conflict-light ds_load_b128
#define HS_STRIDE 72        // 64 + 8 pad
#define W1S_STRIDE 264      // 256 + 8 pad
#define W2S_STRIDE 72       // 64 + 8 pad

typedef __attribute__((ext_vector_type(16))) __bf16 v16bf;
typedef __attribute__((ext_vector_type(8)))  float  v8f;
typedef __attribute__((ext_vector_type(8)))  unsigned short ushort8;

__device__ __forceinline__ unsigned short f32_to_bf16(float f) {
    unsigned int u = __float_as_uint(f);
    unsigned int lsb = (u >> 16) & 1u;
    u += 0x7fffu + lsb;                 // round-to-nearest-even
    return (unsigned short)(u >> 16);
}

__device__ __forceinline__ v16bf pack_bf16(ushort8 lo, ushort8 hi) {
    union { struct { ushort8 lo, hi; } u; v16bf v; } c;
    c.u.lo = lo; c.u.hi = hi;
    return c.v;
}

__device__ __forceinline__ v8f vzero8() {
    v8f v;
    #pragma unroll
    for (int i = 0; i < 8; ++i) v[i] = 0.0f;
    return v;
}

__device__ __forceinline__ v8f wmma_bf16(v16bf a, v16bf b, v8f c) {
    return __builtin_amdgcn_wmma_f32_16x16x32_bf16(
        /*neg_a=*/false, a, /*neg_b=*/false, b,
        /*c_mod=*/(short)0, c, /*reuse_a=*/false, /*reuse_b=*/false);
}

// gfx1250 async copy: global -> LDS, 16B per lane, tracked by ASYNCcnt.
__device__ __forceinline__ void async_ld_b128(unsigned lds_off, const void* gaddr) {
    asm volatile("global_load_async_to_lds_b128 %0, %1, off"
                 :: "v"(lds_off), "v"(gaddr) : "memory");
}
__device__ __forceinline__ void wait_async0() {
    asm volatile("s_wait_asynccnt 0x0" ::: "memory");
}

// ---------------------------------------------------------------------------
// Kernel 1: zero out[] and expert counters
// ---------------------------------------------------------------------------
__global__ __launch_bounds__(256)
void moe_init_kernel(float* __restrict__ out, int* __restrict__ cnt) {
    size_t idx = (size_t)blockIdx.x * blockDim.x + threadIdx.x;
    size_t n = (size_t)NT * DD;
    for (size_t i = idx; i < n; i += (size_t)gridDim.x * blockDim.x) out[i] = 0.0f;
    if (idx < EE) cnt[idx] = 0;
}

// ---------------------------------------------------------------------------
// Kernel 2: convert W1 / W2 to bf16
// ---------------------------------------------------------------------------
__global__ __launch_bounds__(256)
void moe_convert_kernel(const float* __restrict__ W1, const float* __restrict__ W2,
                        unsigned short* __restrict__ w1b, unsigned short* __restrict__ w2b) {
    size_t idx = (size_t)blockIdx.x * blockDim.x + threadIdx.x;
    size_t n = (size_t)EE * HH * DD;
    for (size_t i = idx; i < n; i += (size_t)gridDim.x * blockDim.x) {
        w1b[i] = f32_to_bf16(W1[i]);
        w2b[i] = f32_to_bf16(W2[i]);
    }
}

// ---------------------------------------------------------------------------
// Kernel 3: gating (fp32), top-2 softmax, token routing, x -> bf16
// One wave (32 lanes) per token; 8 waves / block.
// ---------------------------------------------------------------------------
__global__ __launch_bounds__(256)
void moe_gate_kernel(const float* __restrict__ x, const float* __restrict__ Wg,
                     const float* __restrict__ bg,
                     unsigned short* __restrict__ xb,
                     int* __restrict__ cnt, int* __restrict__ tlist,
                     float* __restrict__ plist) {
    __shared__ float wg_s[EE * DD];   // 8 KB
    int tid = threadIdx.x;
    for (int i = tid; i < EE * DD; i += 256) wg_s[i] = Wg[i];
    __syncthreads();

    int wid = tid >> 5, lane = tid & 31;
    int t = blockIdx.x * 8 + wid;

    const float* xp = x + (size_t)t * DD;
    float xv[8];
    #pragma unroll
    for (int j = 0; j < 8; ++j) xv[j] = xp[lane + 32 * j];

    unsigned short* xo = xb + (size_t)t * DD;
    #pragma unroll
    for (int j = 0; j < 8; ++j) xo[lane + 32 * j] = f32_to_bf16(xv[j]);

    float sc[EE];
    #pragma unroll
    for (int e = 0; e < EE; ++e) {
        float s = 0.0f;
        #pragma unroll
        for (int j = 0; j < 8; ++j) s += xv[j] * wg_s[e * DD + lane + 32 * j];
        #pragma unroll
        for (int off = 16; off >= 1; off >>= 1) s += __shfl_xor(s, off, 32);
        sc[e] = s + bg[e];
    }

    if (lane == 0) {
        int i0 = 0; float s0 = sc[0];
        #pragma unroll
        for (int e = 1; e < EE; ++e) if (sc[e] > s0) { s0 = sc[e]; i0 = e; }
        int i1 = -1; float s1 = -3.4e38f;
        #pragma unroll
        for (int e = 0; e < EE; ++e) if (e != i0 && sc[e] > s1) { s1 = sc[e]; i1 = e; }
        float ex = expf(s1 - s0);           // s1 <= s0, no overflow
        float p0 = 1.0f / (1.0f + ex);
        float p1 = 1.0f - p0;
        int pos0 = atomicAdd(&cnt[i0], 1);
        tlist[i0 * NT + pos0] = t; plist[i0 * NT + pos0] = p0;
        int pos1 = atomicAdd(&cnt[i1], 1);
        tlist[i1 * NT + pos1] = t; plist[i1 * NT + pos1] = p1;
    }
}

// ---------------------------------------------------------------------------
// Kernel 4: fused expert FFN with WMMA + double-buffered async weight staging.
//   h = relu(x@W1^T + b1); y = h@W2^T + b2; out[token] += prob * y
// Block: 256 threads (8 waves), tile = 128 tokens of one expert.
// ---------------------------------------------------------------------------
__global__ __launch_bounds__(256, 1)
void moe_expert_kernel(const unsigned short* __restrict__ xb,
                       const unsigned short* __restrict__ w1b,
                       const unsigned short* __restrict__ w2b,
                       const float* __restrict__ b1, const float* __restrict__ b2,
                       const int* __restrict__ cnt, const int* __restrict__ tlist,
                       const float* __restrict__ plist,
                       float* __restrict__ out) {
    __shared__ unsigned short xs[MTILE * XS_STRIDE];          // 66 KB: token tile
    __shared__ unsigned short hs[MTILE * HS_STRIDE];          // 18 KB: relu(h) chunk
    __shared__ unsigned short w1s[2][HCHUNK * W1S_STRIDE];    // 2 x 33 KB: W1 chunk
    __shared__ unsigned short w2s[2][DD * W2S_STRIDE];        // 2 x 36 KB: W2 chunk
    __shared__ int   tok_s[MTILE];
    __shared__ float prob_s[MTILE];

    int e = blockIdx.y;
    int tile0 = blockIdx.x * MTILE;
    int ntok = cnt[e];
    if (tile0 >= ntok) return;                 // uniform early-exit (before barriers)
    int nvalid = ntok - tile0; if (nvalid > MTILE) nvalid = MTILE;

    int tid = threadIdx.x;
    int wid = tid >> 5, lane = tid & 31;
    int lm = lane & 15;                        // row/col within 16
    int kh = (lane >> 4) * 8;                  // bf16 A/B layout K-phase

    const unsigned short* w1e = w1b + (size_t)e * HH * DD;
    const unsigned short* w2e = w2b + (size_t)e * DD * HH;

    // ---- kick off async staging of first weight chunk (no VGPR transit) ----
    {
        #pragma unroll
        for (int i = 0; i < 8; ++i) {          // W1 rows hb..hb+63 are contiguous
            int c = tid + i * 256;             // 2048 chunks of 8 ushorts
            int row = c >> 5, col = (c & 31) * 8;
            async_ld_b128((unsigned)(uintptr_t)&w1s[0][row * W1S_STRIDE + col],
                          w1e + row * DD + col);
        }
        #pragma unroll
        for (int i = 0; i < 8; ++i) {          // W2: 256 rows x 64 cols
            int c = tid + i * 256;
            int row = c >> 3, col = (c & 7) * 8;
            async_ld_b128((unsigned)(uintptr_t)&w2s[0][row * W2S_STRIDE + col],
                          w2e + (size_t)row * HH + col);
        }
    }

    // ---- stage token list, probs, and x tile (bf16) into LDS ----
    for (int i = tid; i < MTILE; i += 256) {
        int tk = -1; float p = 0.0f;
        if (i < nvalid) { tk = tlist[e * NT + tile0 + i]; p = plist[e * NT + tile0 + i]; }
        tok_s[i] = tk; prob_s[i] = p;
    }
    __syncthreads();
    for (int c = tid; c < MTILE * 32; c += 256) {   // 32 x 8-ushort chunks / row
        int r = c >> 5, ch = c & 31;
        ushort8 v = {0, 0, 0, 0, 0, 0, 0, 0};
        int tk = tok_s[r];
        if (tk >= 0) v = *(const ushort8*)(xb + (size_t)tk * DD + ch * 8);
        *(ushort8*)(&xs[r * XS_STRIDE + ch * 8]) = v;
    }
    wait_async0();                             // first weight chunk landed
    __syncthreads();

    v8f yacc[16];
    #pragma unroll
    for (int i = 0; i < 16; ++i) yacc[i] = vzero8();

    int mrow = wid * 16;                       // wave-private 16 token rows
    int rbase = mrow + ((lane >> 4) << 3);     // C-layout row base

    for (int it = 0; it < NCHUNK; ++it) {
        int hb = it * HCHUNK;
        int cur = it & 1, nxt = cur ^ 1;

        // ---- prefetch next weight chunk asynchronously into other buffer ----
        if (it + 1 < NCHUNK) {
            const unsigned short* w1n = w1e + (size_t)(hb + HCHUNK) * DD;
            const unsigned short* w2n = w2e + hb + HCHUNK;
            #pragma unroll
            for (int i = 0; i < 8; ++i) {
                int c = tid + i * 256;
                int row = c >> 5, col = (c & 31) * 8;
                async_ld_b128((unsigned)(uintptr_t)&w1s[nxt][row * W1S_STRIDE + col],
                              w1n + row * DD + col);
            }
            #pragma unroll
            for (int i = 0; i < 8; ++i) {
                int c = tid + i * 256;
                int row = c >> 3, col = (c & 7) * 8;
                async_ld_b128((unsigned)(uintptr_t)&w2s[nxt][row * W2S_STRIDE + col],
                              w2n + (size_t)row * HH + col);
            }
        }

        // ---- phase 1: h_chunk[16 x 64] = x_tile[16 x 256] @ W1_chunk^T ----
        v8f hacc[4];
        #pragma unroll
        for (int nb = 0; nb < 4; ++nb) hacc[nb] = vzero8();

        for (int k0 = 0; k0 < DD; k0 += 32) {
            const unsigned short* ap = &xs[(mrow + lm) * XS_STRIDE + k0 + kh];
            v16bf a = pack_bf16(*(const ushort8*)ap, *(const ushort8*)(ap + 16));
            #pragma unroll
            for (int nb = 0; nb < 4; ++nb) {
                const unsigned short* bp =
                    &w1s[cur][(nb * 16 + lm) * W1S_STRIDE + k0 + kh];
                v16bf b = pack_bf16(*(const ushort8*)bp, *(const ushort8*)(bp + 16));
                hacc[nb] = wmma_bf16(a, b, hacc[nb]);
            }
        }
        // bias + relu -> bf16 into wave-private LDS rows (no barrier needed)
        #pragma unroll
        for (int nb = 0; nb < 4; ++nb) {
            int c = nb * 16 + lm;
            float bias = b1[e * HH + hb + c];
            #pragma unroll
            for (int v = 0; v < 8; ++v) {
                float val = hacc[nb][v] + bias;
                val = fmaxf(val, 0.0f);
                hs[(rbase + v) * HS_STRIDE + c] = f32_to_bf16(val);
            }
        }

        // ---- phase 2: y[16 x 256] += h_chunk[16 x 64] @ W2_chunk^T ----
        for (int k0 = 0; k0 < HCHUNK; k0 += 32) {
            const unsigned short* ap = &hs[(mrow + lm) * HS_STRIDE + k0 + kh];
            v16bf a = pack_bf16(*(const ushort8*)ap, *(const ushort8*)(ap + 16));
            #pragma unroll
            for (int nb = 0; nb < 16; ++nb) {
                const unsigned short* bp =
                    &w2s[cur][(nb * 16 + lm) * W2S_STRIDE + k0 + kh];
                v16bf b = pack_bf16(*(const ushort8*)bp, *(const ushort8*)(bp + 16));
                yacc[nb] = wmma_bf16(a, b, yacc[nb]);
            }
        }

        // publish prefetched chunk; also guards LDS buffer reuse next iter
        wait_async0();
        __syncthreads();
    }

    // ---- epilogue: + b2, scale by combine prob, scatter-add to out ----
    #pragma unroll
    for (int nb = 0; nb < 16; ++nb) {
        int dcol = nb * 16 + lm;
        float bias = b2[e * DD + dcol];
        #pragma unroll
        for (int v = 0; v < 8; ++v) {
            int r = rbase + v;
            if (r < nvalid) {
                int tk = tok_s[r];
                float p = prob_s[r];
                atomicAdd(&out[(size_t)tk * DD + dcol], p * (yacc[nb][v] + bias));
            }
        }
    }
}

// ---------------------------------------------------------------------------
extern "C" void kernel_launch(void* const* d_in, const int* in_sizes, int n_in,
                              void* d_out, int out_size, void* d_ws, size_t ws_size,
                              hipStream_t stream) {
    const float* x  = (const float*)d_in[0];
    const float* Wg = (const float*)d_in[1];
    const float* bg = (const float*)d_in[2];
    const float* W1 = (const float*)d_in[3];
    const float* b1 = (const float*)d_in[4];
    const float* W2 = (const float*)d_in[5];
    const float* b2 = (const float*)d_in[6];
    float* out = (float*)d_out;

    char* ws = (char*)d_ws;
    unsigned short* xb  = (unsigned short*)ws; ws += (size_t)NT * DD * 2;       // 33.5 MB
    unsigned short* w1b = (unsigned short*)ws; ws += (size_t)EE * HH * DD * 2;  // 4.2 MB
    unsigned short* w2b = (unsigned short*)ws; ws += (size_t)EE * DD * HH * 2;  // 4.2 MB
    int*   cnt   = (int*)ws;   ws += 256;
    int*   tlist = (int*)ws;   ws += (size_t)EE * NT * 4;                       // 2 MB
    float* plist = (float*)ws;                                                  // 2 MB

    moe_init_kernel<<<4096, 256, 0, stream>>>(out, cnt);
    moe_convert_kernel<<<2048, 256, 0, stream>>>(W1, W2, w1b, w2b);
    moe_gate_kernel<<<NT / 8, 256, 0, stream>>>(x, Wg, bg, xb, cnt, tlist, plist);
    moe_expert_kernel<<<dim3((NT + MTILE - 1) / MTILE, EE), 256, 0, stream>>>(
        xb, w1b, w2b, b1, b2, cnt, tlist, plist, out);
}
